// BatchGraphConvolution_51848845197335
// MI455X (gfx1250) — compile-verified, matrix operand order checked by reading, not run on hardware
//
#include <hip/hip_runtime.h>

// ---- CDNA5 (gfx1250) wave32 WMMA types ----
typedef _Float16 v4h  __attribute__((ext_vector_type(4)));
typedef _Float16 v8h  __attribute__((ext_vector_type(8)));
typedef _Float16 v16h __attribute__((ext_vector_type(16)));
typedef float    v8f  __attribute__((ext_vector_type(8)));

#define N_NODES 4096
#define F_DIM   32
#define BATCH   64
#define BK      64      // K-step per barrier pair (2 WMMA K-slabs)

// ------------------------------------------------------------------
// Kernel 0: adj fp32 -> fp16 (vectorized, 4 elems/thread)
// ------------------------------------------------------------------
__global__ __launch_bounds__(256) void adj_cvt_kernel(
    const float* __restrict__ src, _Float16* __restrict__ dst, int n4) {
  int i = blockIdx.x * 256 + threadIdx.x;
  if (i < n4) {
    const float4 v = ((const float4*)src)[i];
    v4h h;
    h[0] = (_Float16)v.x; h[1] = (_Float16)v.y;
    h[2] = (_Float16)v.z; h[3] = (_Float16)v.w;
    ((v4h*)dst)[i] = h;
  }
}

// ------------------------------------------------------------------
// Kernel 1: support[b][n][o] = sum_f x[b][n][f] * W[f][o]  (fp32 acc -> fp16)
// 256 threads = 8 rows x 32 outputs; W cached in LDS.
// ------------------------------------------------------------------
__global__ __launch_bounds__(256) void support_kernel(
    const float* __restrict__ x, const float* __restrict__ W,
    _Float16* __restrict__ sup) {
  __shared__ float Wl[F_DIM * F_DIM];
  const int tid = threadIdx.x;
  for (int i = tid; i < F_DIM * F_DIM; i += 256) Wl[i] = W[i];
  __syncthreads();

  const int o  = tid & 31;
  const int rl = tid >> 5;
  const long row = (long)blockIdx.x * 8 + rl;   // 0 .. B*N-1
  const float* xr = x + row * F_DIM;

  float acc = 0.0f;
#pragma unroll
  for (int f = 0; f < F_DIM; ++f) acc = fmaf(xr[f], Wl[f * F_DIM + o], acc);

  sup[row * F_DIM + o] = (_Float16)acc;
}

// ------------------------------------------------------------------
// Kernel 2: out[b] = adj @ support[b] + bias
// f16 WMMA 16x16x32, fp32 accumulation.
// Block = 256 threads (8 wave32s); tile BM=256 x BN=32, BK=64.
// Each wave owns a 32x32 out tile (2x2 accumulators); 8 WMMAs per
// barrier pair (2 K-slabs of 32).
// ------------------------------------------------------------------
__global__ __launch_bounds__(256) void gcn_gemm_kernel(
    const _Float16* __restrict__ adjh,   // [N][N] f16, row-major
    const _Float16* __restrict__ suph,   // [B][N][32] f16
    const float* __restrict__ bias,      // [32]
    float* __restrict__ out) {           // [B][N][32] f32
  __shared__ _Float16 As[256 * BK];      // A tile  [m][k]   32 KB
  __shared__ _Float16 Bt[32 * BK];       // B tile transposed [o][k]  4 KB

  const int tid    = threadIdx.x;
  const int lane   = tid & 31;
  const int wave   = tid >> 5;
  const int lane16 = lane & 15;
  const int laneHi = lane >> 4;          // which K-half this lane holds
  const int mBase  = blockIdx.x * 256;
  const int b      = blockIdx.y;

  const _Float16* aRow = adjh + (size_t)(mBase + tid) * N_NODES; // this thread's adj row
  const _Float16* sB   = suph + (size_t)b * N_NODES * F_DIM;

  // B staging: 32 cols x 64 k = 2048 halves/step, 8 per thread
  const int bk = tid >> 2;               // k row   0..63
  const int bo = (tid & 3) * 8;          // o col   0,8,16,24

  v8f c[2][2] = {};                      // fp32 accumulators (4 x 16x16 tiles)
  const float bias0 = bias[lane16];
  const float bias1 = bias[16 + lane16];

  for (int kb = 0; kb < N_NODES; kb += BK) {
    // ---- global -> regs (128B of A row, 16B of B) ----
    const uint4* ap = (const uint4*)(aRow + kb);
    uint4 a0 = ap[0], a1 = ap[1], a2 = ap[2], a3 = ap[3];
    uint4 a4 = ap[4], a5 = ap[5], a6 = ap[6], a7 = ap[7];
    const v8h bv = *(const v8h*)(sB + (size_t)(kb + bk) * F_DIM + bo);
    if (kb + BK < N_NODES)
      __builtin_prefetch(aRow + kb + BK, 0, 1);            // global_prefetch_b8

    __syncthreads();                                       // LDS consumed by prev iter
    // ---- regs -> LDS ----
    uint4* asp = (uint4*)(As + tid * BK);
    asp[0] = a0; asp[1] = a1; asp[2] = a2; asp[3] = a3;
    asp[4] = a4; asp[5] = a5; asp[6] = a6; asp[7] = a7;
#pragma unroll
    for (int i = 0; i < 8; ++i)
      Bt[(bo + i) * BK + bk] = bv[i];                      // transpose: Bt[o][k]
    __syncthreads();

    // ---- 2 K-slabs of 32; fragments per ISA 7.12.2 16-bit layout ----
    // lane m (or n) = lane16; K elems of slab: half 'laneHi':
    //   frag[0..7] = K = h*8+j, frag[8..15] = K = 16+h*8+j
#pragma unroll
    for (int ks = 0; ks < 2; ++ks) {
      v16h afrag[2], bfrag[2];
#pragma unroll
      for (int mt = 0; mt < 2; ++mt) {
        const _Float16* p = As + (wave * 32 + mt * 16 + lane16) * BK +
                            ks * 32 + laneHi * 8;
        v8h lo = *(const v8h*)p;
        v8h hi = *(const v8h*)(p + 16);
        afrag[mt] = __builtin_shufflevector(lo, hi,
            0,1,2,3,4,5,6,7,8,9,10,11,12,13,14,15);
      }
#pragma unroll
      for (int nt = 0; nt < 2; ++nt) {
        const _Float16* p = Bt + (nt * 16 + lane16) * BK +
                            ks * 32 + laneHi * 8;
        v8h lo = *(const v8h*)p;
        v8h hi = *(const v8h*)(p + 16);
        bfrag[nt] = __builtin_shufflevector(lo, hi,
            0,1,2,3,4,5,6,7,8,9,10,11,12,13,14,15);
      }
#pragma unroll
      for (int mt = 0; mt < 2; ++mt)
#pragma unroll
        for (int nt = 0; nt < 2; ++nt)
          c[mt][nt] = __builtin_amdgcn_wmma_f32_16x16x32_f16(
              false, afrag[mt], false, bfrag[nt],
              (short)0, c[mt][nt], false, false);
    }
  }

  // ---- epilogue: C/D layout (VGPR r: lanes 0-15 -> M=r, lanes 16-31 -> M=8+r) ----
#pragma unroll
  for (int mt = 0; mt < 2; ++mt) {
#pragma unroll
    for (int nt = 0; nt < 2; ++nt) {
      const float bv  = nt ? bias1 : bias0;
      const int   col = nt * 16 + lane16;
#pragma unroll
      for (int r = 0; r < 8; ++r) {
        const int row = mBase + wave * 32 + mt * 16 + laneHi * 8 + r;
        out[((size_t)b * N_NODES + row) * F_DIM + col] = c[mt][nt][r] + bv;
      }
    }
  }
}

// ------------------------------------------------------------------
extern "C" void kernel_launch(void* const* d_in, const int* in_sizes, int n_in,
                              void* d_out, int out_size, void* d_ws, size_t ws_size,
                              hipStream_t stream) {
  const float* x    = (const float*)d_in[0];   // [64][4096][32]
  const float* adj  = (const float*)d_in[1];   // [4096][4096]
  const float* W    = (const float*)d_in[2];   // [32][32]
  const float* bias = (const float*)d_in[3];   // [32]
  float* out        = (float*)d_out;           // [64][4096][32]

  _Float16* adjh = (_Float16*)d_ws;                                   // 32 MB
  _Float16* suph = (_Float16*)((char*)d_ws +
                     (size_t)N_NODES * N_NODES * sizeof(_Float16));   // +16 MB

  // 1) adj fp32 -> fp16
  const int n4 = N_NODES * N_NODES / 4;
  adj_cvt_kernel<<<n4 / 256, 256, 0, stream>>>(adj, adjh, n4);

  // 2) support = x @ W  (fp16 out)
  support_kernel<<<(BATCH * N_NODES) / 8, 256, 0, stream>>>(x, W, suph);

  // 3) out = adj @ support + bias  (WMMA f16 -> f32)
  dim3 grid(N_NODES / 256, BATCH);
  gcn_gemm_kernel<<<grid, 256, 0, stream>>>(adjh, suph, bias, out);
}
